// Gemma4Router_46969762349449
// MI455X (gfx1250) — compile-verified, matrix operand order checked by reading, not run on hardware
//
#include <hip/hip_runtime.h>

typedef __attribute__((ext_vector_type(16))) _Float16 v16h;
typedef __attribute__((ext_vector_type(8)))  float    v8f;
typedef __attribute__((ext_vector_type(2)))  float    v2f;

#define HIDDEN   2048
#define NEXP     128
#define TOPK     4
#define KCH      (HIDDEN / 32)   // 64 K-chunks of 32
#define NTW      4               // N-tiles per wave (64 experts)
#define TPB      32              // tokens per block (2 groups of 16)
#define LPAD     132             // LDS logits row stride (floats)

// ---------------------------------------------------------------------------
// Kernel 0: fold scale[k] * hidden^-0.5 into W, convert to f16, pre-swizzle
// into the per-lane WMMA B-fragment layout:
//   wsb[((kc*8 + nt)*32 + lane)*16 + j]
//     = f16( W[nt*16 + (lane&15)][ kc*32 + ((lane&16)?8:0) + (j<8 ? j : j+8) ]
//            * scale[k] * hscale )
// One contiguous 32-byte load per lane per B fragment in the main kernel.
// ---------------------------------------------------------------------------
__global__ void router_prep_w(const float* __restrict__ W,
                              const float* __restrict__ scale,
                              _Float16* __restrict__ wsb) {
  int tid  = blockIdx.x * blockDim.x + threadIdx.x;  // 0 .. NEXP*HIDDEN-1
  int j    = tid & 15;
  int lane = (tid >> 4) & 31;
  int nt   = (tid >> 9) & 7;
  int kc   = tid >> 12;
  int e = nt * 16 + (lane & 15);
  int k = kc * 32 + ((lane & 16) ? 8 : 0) + (j < 8 ? j : j + 8);
  const float hs = 0.022097086912079608f;  // 1/sqrt(2048)
  wsb[tid] = (_Float16)(W[(size_t)e * HIDDEN + k] * scale[k] * hs);
}

// ---------------------------------------------------------------------------
// Main kernel: 128 threads = 4 waves; block covers 32 tokens x 128 experts.
//   wave w: token group g = w>>1 (16 tokens), expert half h = w&1 (64 experts)
// Pass 1: each wave computes inv_rms for 8 distinct tokens -> LDS sinv[32].
// Pass 2: WMMA f16 GEMM, A = f16(x) raw; inv_rms folded at logit spill.
// Pass 3: per-token top-4 (softmax Z cancels after renormalization).
// ---------------------------------------------------------------------------
__global__ void __launch_bounds__(128) router_main(
    const float*    __restrict__ x,
    const _Float16* __restrict__ wsb,
    const float*    __restrict__ pes,      // per_expert_scale[128]
    int*            __restrict__ out_idx,  // [n_tokens][4]
    float*          __restrict__ out_w) {  // [n_tokens][4]
  __shared__ float slog[TPB * LPAD];       // 32 x 128 logits (padded)
  __shared__ float sinv[TPB];              // inv_rms per block-local token

  const int lane     = threadIdx.x & 31;
  const int wave     = threadIdx.x >> 5;
  const int g        = wave >> 1;          // token group within block
  const int h        = wave & 1;           // expert half
  const int blockTok = blockIdx.x * TPB;
  const int tokBase  = blockTok + g * 16;  // this wave's 16 tokens

  // ---------------- Pass 1: inv_rms for 8 tokens per wave -----------------
  // Packed f32 squares -> v_pk_fma_f32 (halves VALU issue vs scalar fma).
  for (int tt = 0; tt < 8; ++tt) {
    const int ltok = wave * 8 + tt;        // block-local token 0..31
    const float* xp = x + (size_t)(blockTok + ltok) * HIDDEN + lane * 4;
    v2f sv = (v2f){0.0f, 0.0f};
#pragma unroll
    for (int i = 0; i < 16; ++i) {
      float4 v = *(const float4*)(xp + i * 128);
      v2f p0 = (v2f){v.x, v.y};
      v2f p1 = (v2f){v.z, v.w};
      sv += p0 * p0;
      sv += p1 * p1;
    }
    float ss = sv[0] + sv[1];
    ss += __shfl_xor(ss, 16);
    ss += __shfl_xor(ss, 8);
    ss += __shfl_xor(ss, 4);
    ss += __shfl_xor(ss, 2);
    ss += __shfl_xor(ss, 1);
    if (lane == 0) sinv[ltok] = rsqrtf(ss * (1.0f / HIDDEN) + 1e-6f);
  }

  // ---------------- Pass 2: logits via v_wmma_f32_16x16x32_f16 ------------
  v8f acc0 = (v8f){0, 0, 0, 0, 0, 0, 0, 0};
  v8f acc1 = (v8f){0, 0, 0, 0, 0, 0, 0, 0};
  v8f acc2 = (v8f){0, 0, 0, 0, 0, 0, 0, 0};
  v8f acc3 = (v8f){0, 0, 0, 0, 0, 0, 0, 0};

  const int    row   = lane & 15;
  const int    koff  = (lane & 16) ? 8 : 0;
  const float* xrow  = x + (size_t)(tokBase + row) * HIDDEN + koff;
  const _Float16* bb = wsb + (size_t)(h * NTW) * 512 + lane * 16;

#pragma clang loop unroll(disable)
  for (int kc = 0; kc < KCH; ++kc) {
    const float* ap = xrow + kc * 32;
    if (kc + 1 < KCH) {
      // WGP-scope prefetch of next x chunk (scope field 0 = WGP: fills all
      // cache levels). The builtin can only reach SE/SYS scope.
      const float* pf = ap + 32;
      asm volatile("global_prefetch_b8 %0, off" :: "v"(pf) : "memory");
    }

    // Issue ALL loads for this chunk up front (4x b128 of x, 8x b128 of B),
    // then fence so the scheduler can't sink B loads between the WMMAs
    // (which would serialize them through a single reused register octet).
    float4 f0 = *(const float4*)(ap + 0);
    float4 f1 = *(const float4*)(ap + 4);
    float4 f2 = *(const float4*)(ap + 16);
    float4 f3 = *(const float4*)(ap + 20);

    const _Float16* bp = bb + (size_t)kc * 4096;   // 8 tiles * 512 f16 / chunk
    v16h b0 = *(const v16h*)(bp + 0 * 512);
    v16h b1 = *(const v16h*)(bp + 1 * 512);
    v16h b2 = *(const v16h*)(bp + 2 * 512);
    v16h b3 = *(const v16h*)(bp + 3 * 512);

    asm volatile("" ::: "memory");   // scheduling fence: loads stay clustered

    v16h a;
    a[0]  = (_Float16)f0.x;  a[1]  = (_Float16)f0.y;
    a[2]  = (_Float16)f0.z;  a[3]  = (_Float16)f0.w;
    a[4]  = (_Float16)f1.x;  a[5]  = (_Float16)f1.y;
    a[6]  = (_Float16)f1.z;  a[7]  = (_Float16)f1.w;
    a[8]  = (_Float16)f2.x;  a[9]  = (_Float16)f2.y;
    a[10] = (_Float16)f2.z;  a[11] = (_Float16)f2.w;
    a[12] = (_Float16)f3.x;  a[13] = (_Float16)f3.y;
    a[14] = (_Float16)f3.z;  a[15] = (_Float16)f3.w;

    acc0 = __builtin_amdgcn_wmma_f32_16x16x32_f16(false, a, false, b0,
                                                  (short)0, acc0, false, false);
    acc1 = __builtin_amdgcn_wmma_f32_16x16x32_f16(false, a, false, b1,
                                                  (short)0, acc1, false, false);
    acc2 = __builtin_amdgcn_wmma_f32_16x16x32_f16(false, a, false, b2,
                                                  (short)0, acc2, false, false);
    acc3 = __builtin_amdgcn_wmma_f32_16x16x32_f16(false, a, false, b3,
                                                  (short)0, acc3, false, false);
  }

  __syncthreads();  // sinv visible to all waves

  // ---------------- Spill logits (scaled by inv_rms) to LDS ---------------
  // C/D layout: VGPR j -> (M = j + (lane>=16 ? 8 : 0), N = lane & 15).
  {
    const int mbase = (lane & 16) ? 8 : 0;
    float rinv[8];
#pragma unroll
    for (int j = 0; j < 8; ++j) rinv[j] = sinv[g * 16 + mbase + j];

    float* ls = slog + (size_t)(g * 16 + mbase) * LPAD + h * 64 + (lane & 15);
#pragma unroll
    for (int j = 0; j < 8; ++j) {
      ls[j * LPAD + 0]  = acc0[j] * rinv[j];
      ls[j * LPAD + 16] = acc1[j] * rinv[j];
      ls[j * LPAD + 32] = acc2[j] * rinv[j];
      ls[j * LPAD + 48] = acc3[j] * rinv[j];
    }
  }
  __syncthreads();

  // ---------------- Top-4 + renormalize (softmax Z cancels) ---------------
  if (lane < 8) {
    const int lt = wave * 8 + lane;        // block-local token
    const float* lp = slog + (size_t)lt * LPAD;
    float b0 = -3.4e38f, b1 = -3.4e38f, b2 = -3.4e38f, b3 = -3.4e38f;
    int   i0 = 0, i1 = 0, i2 = 0, i3 = 0;
    for (int e = 0; e < NEXP; ++e) {
      float v = lp[e];
      if (v > b3) {                        // strict > => stable (low idx wins)
        if (v > b0)      { b3 = b2; i3 = i2; b2 = b1; i2 = i1;
                           b1 = b0; i1 = i0; b0 = v;  i0 = e; }
        else if (v > b1) { b3 = b2; i3 = i2; b2 = b1; i2 = i1;
                           b1 = v;  i1 = e; }
        else if (v > b2) { b3 = b2; i3 = i2; b2 = v;  i2 = e; }
        else             { b3 = v;  i3 = e; }
      }
    }
    float w0 = 1.0f;
    float w1 = __expf(b1 - b0);
    float w2 = __expf(b2 - b0);
    float w3 = __expf(b3 - b0);
    float inv = 1.0f / (w0 + w1 + w2 + w3);

    const int t = blockTok + lt;
    out_idx[t * TOPK + 0] = i0;
    out_idx[t * TOPK + 1] = i1;
    out_idx[t * TOPK + 2] = i2;
    out_idx[t * TOPK + 3] = i3;
    out_w[t * TOPK + 0] = w0 * inv * pes[i0];
    out_w[t * TOPK + 1] = w1 * inv * pes[i1];
    out_w[t * TOPK + 2] = w2 * inv * pes[i2];
    out_w[t * TOPK + 3] = w3 * inv * pes[i3];
  }
}

// ---------------------------------------------------------------------------
extern "C" void kernel_launch(void* const* d_in, const int* in_sizes, int n_in,
                              void* d_out, int out_size, void* d_ws, size_t ws_size,
                              hipStream_t stream) {
  (void)n_in; (void)out_size; (void)ws_size;
  const float* x     = (const float*)d_in[0];  // [n_tokens, 2048]
  const float* scale = (const float*)d_in[1];  // [2048]
  const float* W     = (const float*)d_in[2];  // [128, 2048]
  const float* pes   = (const float*)d_in[3];  // [128]

  const int n_tokens = in_sizes[0] / HIDDEN;

  _Float16* wsb = (_Float16*)d_ws;             // 128*2048 f16 = 512 KB

  router_prep_w<<<(NEXP * HIDDEN) / 256, 256, 0, stream>>>(W, scale, wsb);

  int*   oi = (int*)d_out;                               // top_k_index
  float* ow = (float*)d_out + (size_t)n_tokens * TOPK;   // top_k_weights

  router_main<<<n_tokens / TPB, 128, 0, stream>>>(x, wsb, pes, oi, ow);
}
